// WanSelfAttention_56100862820961
// MI455X (gfx1250) — matrix operand order repaired and checked by reading, not compiled, not run on hardware
//
#include <hip/hip_runtime.h>
#include <hip/hip_bf16.h>

// ---------------- problem constants (compile-time in the reference) ----------
#define DIM_C   1536
#define NH_C    12
#define HD_C    128
#define S_C     1560           // SEQ = 1*30*52
#define F_C     1
#define H_C     30
#define W_C     52
#define GI_C    5
#define START_C 7800           // GI * SEQ
#define LTOT_C  9360           // START + SEQ (full cache attended)
#define CF_C    22             // c=64; cf = 64 - 2*(64/3)
#define CH_C    21
#define EPS_C   1e-6f
#define QSCALE_C 0.08838834764831845f   // 1/sqrt(128), folded into q

typedef __bf16 bf16_t;
typedef __attribute__((ext_vector_type(16))) __bf16 v16bf;
typedef __attribute__((ext_vector_type(8)))  float  v8f;

__device__ __forceinline__ bf16_t f2bf(float f) {
  union { float f; unsigned u; } x; x.f = f;
  unsigned r = x.u + 0x7FFFu + ((x.u >> 16) & 1u);   // round-to-nearest-even
  unsigned short h = (unsigned short)(r >> 16);
  return __builtin_bit_cast(bf16_t, h);
}

__device__ __forceinline__ v8f zerov8() {
  v8f z = {0.f, 0.f, 0.f, 0.f, 0.f, 0.f, 0.f, 0.f};
  return z;
}

__device__ __forceinline__ v8f wmma_bf16(v16bf a, v16bf b, v8f c) {
  // D = A(16x32) * B(32x16) + C, fp32 accumulate
  return __builtin_amdgcn_wmma_f32_16x16x32_bf16(false, a, false, b, (short)0, c,
                                                 false, false);
}

// A-operand frag: 16(M) x 32(K) bf16; memory row-major, K contiguous.
// Lane l: row = l&15 (clamped), K halves { (l>>4)*8 .. +7 } and { +16 .. +23 }.
__device__ __forceinline__ v16bf load_a_frag(const bf16_t* tile, int ld, int lane, int rmax) {
  int r = lane & 15; if (r > rmax) r = rmax; if (r < 0) r = 0;
  int kh = (lane >> 4) << 3;                 // 0 or 8
  const bf16_t* p = tile + (size_t)r * ld + kh;
  union { v16bf v; uint4 q[2]; } f;
  f.q[0] = *(const uint4*)(p);
  f.q[1] = *(const uint4*)(p + 16);
  return f.v;
}

// B-operand frag: 32(K) x 16(N) bf16, sourced from memory laid out [n][k]
// (K contiguous per n-row). Lane l: n = l&15 (clamped), K halves { (l>>4)*8 .. +7 } and { +16 .. +23 }.
__device__ __forceinline__ v16bf load_b_frag(const bf16_t* tile, int ld, int lane, int rmax) {
  int r = lane & 15; if (r > rmax) r = rmax; if (r < 0) r = 0;
  int kh = (lane >> 4) << 3;                 // 0 or 8
  const bf16_t* p = tile + (size_t)r * ld + kh;
  union { v16bf v; uint4 q[2]; } f;
  f.q[0] = *(const uint4*)(p);
  f.q[1] = *(const uint4*)(p + 16);
  return f.v;
}

// ---------------------------------------------------------------- casts -----
__global__ void cast_f32_bf16_kernel(const float* __restrict__ in,
                                     bf16_t* __restrict__ out, long long n) {
  long long i = (long long)blockIdx.x * blockDim.x + threadIdx.x;
  long long stride = (long long)gridDim.x * blockDim.x;
  for (; i < n; i += stride) out[i] = f2bf(in[i]);
}

// v_cache rows [0, rows) -> vT[(n*128+d)][key]  (key contiguous)
__global__ void cast_vT_kernel(const float* __restrict__ vc,
                               bf16_t* __restrict__ vT, long long rows) {
  long long n = rows * DIM_C;
  long long i = (long long)blockIdx.x * blockDim.x + threadIdx.x;
  long long stride = (long long)gridDim.x * blockDim.x;
  for (; i < n; i += stride) {
    long long key = i / DIM_C;
    long long d   = i - key * DIM_C;
    vT[d * LTOT_C + key] = f2bf(vc[i]);
  }
}

// ------------------------------------------------- bf16 NT GEMM (WMMA) ------
// C[M,N] = A[M,K] * B[N,K]^T + bias.
// Block = 8 waves; each wave owns a 32x64 tile (2 A-frags reuse 4 B-frags:
// 8 WMMA per 12 b128 loads per k-step). Block covers 256 rows x 64 cols.
__global__ __launch_bounds__(256) void gemm_nt_bf16_kernel(
    const bf16_t* __restrict__ A, const bf16_t* __restrict__ B,
    const float* __restrict__ bias, float* __restrict__ C,
    int M, int N, int K) {
  int wave = threadIdx.x >> 5;
  int lane = threadIdx.x & 31;
  int m0 = blockIdx.x * 256 + wave * 32;
  int n0 = blockIdx.y * 64;
  if (m0 >= M) return;
  int m1 = (m0 + 16 < M) ? (m0 + 16) : m0;   // second row-tile (clamped)
  int rmax0 = M - 1 - m0; if (rmax0 > 15) rmax0 = 15;
  int rmax1 = M - 1 - m1; if (rmax1 > 15) rmax1 = 15;

  v8f acc[2][4];
#pragma unroll
  for (int t = 0; t < 2; ++t)
#pragma unroll
    for (int j = 0; j < 4; ++j) acc[t][j] = zerov8();

  const bf16_t* arow0 = A + (size_t)m0 * K;
  const bf16_t* arow1 = A + (size_t)m1 * K;
  for (int kb = 0; kb < K; kb += 32) {
    // speculative prefetch two k-steps ahead (global_prefetch_b8; dropped if OOB)
    __builtin_prefetch(arow0 + kb + 64, 0, 1);
    __builtin_prefetch(B + (size_t)(n0 + (lane & 63)) * K + kb + 64, 0, 1);

    v16bf a0 = load_a_frag(arow0 + kb, K, lane, rmax0);
    v16bf a1 = load_a_frag(arow1 + kb, K, lane, rmax1);
    v16bf b0 = load_b_frag(B + (size_t)(n0 +  0) * K + kb, K, lane, 15);
    v16bf b1 = load_b_frag(B + (size_t)(n0 + 16) * K + kb, K, lane, 15);
    v16bf b2 = load_b_frag(B + (size_t)(n0 + 32) * K + kb, K, lane, 15);
    v16bf b3 = load_b_frag(B + (size_t)(n0 + 48) * K + kb, K, lane, 15);
    acc[0][0] = wmma_bf16(a0, b0, acc[0][0]);
    acc[1][0] = wmma_bf16(a1, b0, acc[1][0]);
    acc[0][1] = wmma_bf16(a0, b1, acc[0][1]);
    acc[1][1] = wmma_bf16(a1, b1, acc[1][1]);
    acc[0][2] = wmma_bf16(a0, b2, acc[0][2]);
    acc[1][2] = wmma_bf16(a1, b2, acc[1][2]);
    acc[0][3] = wmma_bf16(a0, b3, acc[0][3]);
    acc[1][3] = wmma_bf16(a1, b3, acc[1][3]);
  }
  // C/D layout: lane&15 = N-col, VGPR r = row r (lanes 0-15) / r+8 (lanes 16-31)
  int rowoff = (lane >> 4) << 3;
#pragma unroll
  for (int t = 0; t < 2; ++t) {
    int mb = (t == 0) ? m0 : m1;
#pragma unroll
    for (int j = 0; j < 4; ++j) {
      int col = n0 + j * 16 + (lane & 15);
      float bv = bias ? bias[col] : 0.f;
#pragma unroll
      for (int r = 0; r < 8; ++r) {
        int row = mb + rowoff + r;
        if (row < M) C[(size_t)row * N + col] = acc[t][j][r] + bv;
      }
    }
  }
}

// ------------------- RMSNorm (over DIM) + 3D-RoPE + cache merge --------------
__global__ __launch_bounds__(256) void qkv_post_kernel(
    const float* __restrict__ Qp, const float* __restrict__ Kp,
    const float* __restrict__ Vp, const float* __restrict__ nqw,
    const float* __restrict__ nkw, const float* __restrict__ freqs,
    bf16_t* __restrict__ qb, bf16_t* __restrict__ kfull,
    bf16_t* __restrict__ vT) {
  int s = blockIdx.x;
  int tid = threadIdx.x;
  __shared__ float cs_[64], sn_[64];
  __shared__ float red[256];

  int wx = s % W_C;
  int hy = (s / W_C) % H_C;
  int fz = s / (W_C * H_C);
  if (tid < 64) {
    int j = tid;
    int pos = (j < CF_C) ? (GI_C * F_C + fz) : ((j < CF_C + CH_C) ? hy : wx);
    cs_[j] = freqs[((size_t)pos * 64 + j) * 2];
    sn_[j] = freqs[((size_t)pos * 64 + j) * 2 + 1];
  }

  const float* qrow = Qp + (size_t)s * DIM_C;
  const float* krow = Kp + (size_t)s * DIM_C;
  float sq = 0.f, sk = 0.f;
  for (int i = tid; i < DIM_C; i += 256) {
    float a = qrow[i]; sq += a * a;
    float b = krow[i]; sk += b * b;
  }
  red[tid] = sq; __syncthreads();
  for (int st = 128; st > 0; st >>= 1) { if (tid < st) red[tid] += red[tid + st]; __syncthreads(); }
  float ssq = red[0]; __syncthreads();
  red[tid] = sk; __syncthreads();
  for (int st = 128; st > 0; st >>= 1) { if (tid < st) red[tid] += red[tid + st]; __syncthreads(); }
  float ssk = red[0]; __syncthreads();

  float scq = rsqrtf(ssq / DIM_C + EPS_C) * QSCALE_C;  // attention scale folded
  float sck = rsqrtf(ssk / DIM_C + EPS_C);

  for (int idx = tid; idx < NH_C * 64; idx += 256) {
    int n = idx >> 6, j = idx & 63;
    int d0 = n * HD_C + 2 * j;
    float c = cs_[j], sn = sn_[j];
    float qr = qrow[d0]     * scq * nqw[d0];
    float qi = qrow[d0 + 1] * scq * nqw[d0 + 1];
    qb[(size_t)s * DIM_C + d0]     = f2bf(qr * c - qi * sn);
    qb[(size_t)s * DIM_C + d0 + 1] = f2bf(qr * sn + qi * c);
    float kr = krow[d0]     * sck * nkw[d0];
    float ki = krow[d0 + 1] * sck * nkw[d0 + 1];
    size_t ko = (size_t)(START_C + s) * DIM_C + d0;
    kfull[ko]     = f2bf(kr * c - ki * sn);
    kfull[ko + 1] = f2bf(kr * sn + ki * c);
  }
  const float* vrow = Vp + (size_t)s * DIM_C;
  for (int i = tid; i < DIM_C; i += 256)
    vT[(size_t)i * LTOT_C + START_C + s] = f2bf(vrow[i]);
}

// ------------------------------- flash attention ----------------------------
// grid = (ceil(S/16), NH); block = 256 (8 waves). Each wave: 32 keys / iter.
__global__ __launch_bounds__(256) void attn_kernel(
    const bf16_t* __restrict__ qb, const bf16_t* __restrict__ kfull,
    const bf16_t* __restrict__ vT, bf16_t* __restrict__ ob) {
  const int L = LTOT_C;
  int m0 = blockIdx.x * 16;
  int head = blockIdx.y;
  int wave = threadIdx.x >> 5;
  int lane = threadIdx.x & 31;

  __shared__ alignas(16) bf16_t Psh[8][16][32];
  __shared__ float Msh[8][16], Lsh[8][16];
  __shared__ float Mrow[16], Ltot[16];
  __shared__ float Osh[16][HD_C];

  for (int i = threadIdx.x; i < 16 * HD_C; i += 256) ((float*)Osh)[i] = 0.f;
  __syncthreads();

  // Q fragments (rows m0..m0+15, d = 0..127 in 4 chunks of 32)
  int rmaxq = S_C - 1 - m0; if (rmaxq > 15) rmaxq = 15;
  const bf16_t* qtile = qb + (size_t)m0 * DIM_C + head * HD_C;
  v16bf qf0 = load_a_frag(qtile +  0, DIM_C, lane, rmaxq);
  v16bf qf1 = load_a_frag(qtile + 32, DIM_C, lane, rmaxq);
  v16bf qf2 = load_a_frag(qtile + 64, DIM_C, lane, rmaxq);
  v16bf qf3 = load_a_frag(qtile + 96, DIM_C, lane, rmaxq);

  v8f oacc[8];
#pragma unroll
  for (int j = 0; j < 8; ++j) oacc[j] = zerov8();
  float mrow[8], lrow[8];
#pragma unroll
  for (int r = 0; r < 8; ++r) { mrow[r] = -1e30f; lrow[r] = 0.f; }

  int prow = (lane >> 4) << 3;   // C-layout row base for this half-wave

  for (int kb = wave * 32; kb < L; kb += 256) {
    bool t1valid = (kb + 16) < L;
    v8f sc0 = zerov8(), sc1 = zerov8();
    {
      int rm0 = L - 1 - kb;        if (rm0 > 15) rm0 = 15;
      const bf16_t* kt0 = kfull + (size_t)kb * DIM_C + head * HD_C;
      __builtin_prefetch(kt0 + 256 * DIM_C, 0, 1);   // next key-block for this wave
      sc0 = wmma_bf16(qf0, load_b_frag(kt0 +  0, DIM_C, lane, rm0), sc0);
      sc0 = wmma_bf16(qf1, load_b_frag(kt0 + 32, DIM_C, lane, rm0), sc0);
      sc0 = wmma_bf16(qf2, load_b_frag(kt0 + 64, DIM_C, lane, rm0), sc0);
      sc0 = wmma_bf16(qf3, load_b_frag(kt0 + 96, DIM_C, lane, rm0), sc0);
      if (t1valid) {
        int rm1 = L - 1 - (kb + 16); if (rm1 > 15) rm1 = 15;
        const bf16_t* kt1 = kt0 + (size_t)16 * DIM_C;
        sc1 = wmma_bf16(qf0, load_b_frag(kt1 +  0, DIM_C, lane, rm1), sc1);
        sc1 = wmma_bf16(qf1, load_b_frag(kt1 + 32, DIM_C, lane, rm1), sc1);
        sc1 = wmma_bf16(qf2, load_b_frag(kt1 + 64, DIM_C, lane, rm1), sc1);
        sc1 = wmma_bf16(qf3, load_b_frag(kt1 + 96, DIM_C, lane, rm1), sc1);
      }
    }
    // online softmax (row lives across one 16-lane half per VGPR)
    int key0 = kb + (lane & 15);
    bool ob0 = key0 >= L, ob1 = (key0 + 16) >= L;
    float alpha[8];
#pragma unroll
    for (int r = 0; r < 8; ++r) {
      float v0 = ob0 ? -1e30f : sc0[r];
      float v1 = ob1 ? -1e30f : sc1[r];
      float bm = fmaxf(v0, v1);
      bm = fmaxf(bm, __shfl_xor(bm, 1, 16));
      bm = fmaxf(bm, __shfl_xor(bm, 2, 16));
      bm = fmaxf(bm, __shfl_xor(bm, 4, 16));
      bm = fmaxf(bm, __shfl_xor(bm, 8, 16));
      float mn = fmaxf(mrow[r], bm);
      alpha[r] = __expf(mrow[r] - mn);
      float p0 = __expf(v0 - mn);
      float p1 = __expf(v1 - mn);
      sc0[r] = p0; sc1[r] = p1;
      float ps = p0 + p1;
      ps += __shfl_xor(ps, 1, 16);
      ps += __shfl_xor(ps, 2, 16);
      ps += __shfl_xor(ps, 4, 16);
      ps += __shfl_xor(ps, 8, 16);
      lrow[r] = lrow[r] * alpha[r] + ps;
      mrow[r] = mn;
    }
#pragma unroll
    for (int j = 0; j < 8; ++j)
#pragma unroll
      for (int r = 0; r < 8; ++r) oacc[j][r] = oacc[j][r] * alpha[r];

    // stage P (C-layout) -> LDS -> reload in A-layout
#pragma unroll
    for (int r = 0; r < 8; ++r) {
      Psh[wave][prow + r][(lane & 15)]      = f2bf(sc0[r]);
      Psh[wave][prow + r][16 + (lane & 15)] = f2bf(sc1[r]);
    }
    union { v16bf v; uint4 q[2]; } pf;
    {
      const bf16_t* pp = &Psh[wave][lane & 15][(lane >> 4) << 3];
      pf.q[0] = *(const uint4*)(pp);
      pf.q[1] = *(const uint4*)(pp + 16);
    }
    // O += P(16x32) * V(32 x 128)
#pragma unroll
    for (int j = 0; j < 8; ++j) {
      const bf16_t* vtile = vT + (size_t)(head * HD_C + j * 16) * L + kb;
      oacc[j] = wmma_bf16(pf.v, load_b_frag(vtile, L, lane, 15), oacc[j]);
    }
  }

  // cross-wave merge
  if ((lane & 15) == 0) {
#pragma unroll
    for (int r = 0; r < 8; ++r) { Msh[wave][prow + r] = mrow[r]; Lsh[wave][prow + r] = lrow[r]; }
  }
  __syncthreads();
  if (threadIdx.x < 16) {
    int row = threadIdx.x;
    float M = -1e30f;
    for (int w2 = 0; w2 < 8; ++w2) M = fmaxf(M, Msh[w2][row]);
    float lt = 0.f;
    for (int w2 = 0; w2 < 8; ++w2) lt += __expf(Msh[w2][row] - M) * Lsh[w2][row];
    Mrow[row] = M;
    Ltot[row] = (lt > 0.f) ? lt : 1.f;
  }
  __syncthreads();
#pragma unroll
  for (int r = 0; r < 8; ++r) {
    float wgt = __expf(mrow[r] - Mrow[prow + r]);
#pragma unroll
    for (int j = 0; j < 8; ++j)
      atomicAdd(&Osh[prow + r][j * 16 + (lane & 15)], oacc[j][r] * wgt);
  }
  __syncthreads();
  for (int idx = threadIdx.x; idx < 16 * HD_C; idx += 256) {
    int row = idx >> 7, col = idx & 127;
    int srow = m0 + row;
    if (srow < S_C)
      ob[(size_t)srow * DIM_C + head * HD_C + col] = f2bf(Osh[row][col] / Ltot[row]);
  }
}

// ------------------------------------------------------------- launcher -----
static inline int gridFor(long long n) {
  long long b = (n + 255) / 256;
  if (b > 4096) b = 4096;
  if (b < 1) b = 1;
  return (int)b;
}

extern "C" void kernel_launch(void* const* d_in, const int* in_sizes, int n_in,
                              void* d_out, int out_size, void* d_ws, size_t ws_size,
                              hipStream_t stream) {
  const float* x    = (const float*)d_in[0];
  const float* wq   = (const float*)d_in[1];
  const float* bq   = (const float*)d_in[2];
  const float* wk   = (const float*)d_in[3];
  const float* bk   = (const float*)d_in[4];
  const float* wv   = (const float*)d_in[5];
  const float* bv   = (const float*)d_in[6];
  const float* wo   = (const float*)d_in[7];
  const float* bo   = (const float*)d_in[8];
  const float* nqw  = (const float*)d_in[9];
  const float* nkw  = (const float*)d_in[10];
  const float* freqs   = (const float*)d_in[11];
  const float* k_cache = (const float*)d_in[12];
  const float* v_cache = (const float*)d_in[13];
  float* out = (float*)d_out;

  // workspace carve (256B aligned, ~120 MB total)
  char* ws = (char*)d_ws;
  size_t off = 0;
  auto carve = [&](size_t bytes) -> void* {
    void* p = ws + off;
    off += (bytes + 255) & ~(size_t)255;
    return p;
  };
  const size_t SD  = (size_t)S_C * DIM_C;
  const size_t DD  = (size_t)DIM_C * DIM_C;
  const size_t LD  = (size_t)LTOT_C * DIM_C;
  bf16_t* xb  = (bf16_t*)carve(SD * 2);
  bf16_t* wqb = (bf16_t*)carve(DD * 2);
  bf16_t* wkb = (bf16_t*)carve(DD * 2);
  bf16_t* wvb = (bf16_t*)carve(DD * 2);
  bf16_t* wob = (bf16_t*)carve(DD * 2);
  float*  Qp  = (float*)carve(SD * 4);
  float*  Kp  = (float*)carve(SD * 4);
  float*  Vp  = (float*)carve(SD * 4);
  bf16_t* qb  = (bf16_t*)carve(SD * 2);
  bf16_t* kf  = (bf16_t*)carve(LD * 2);
  bf16_t* vT  = (bf16_t*)carve(LD * 2);
  bf16_t* obf = (bf16_t*)carve(SD * 2);   // placed after vT: absorbs tail over-read

  // 1) casts to bf16
  cast_f32_bf16_kernel<<<gridFor(SD), 256, 0, stream>>>(x, xb, (long long)SD);
  cast_f32_bf16_kernel<<<gridFor(DD), 256, 0, stream>>>(wq, wqb, (long long)DD);
  cast_f32_bf16_kernel<<<gridFor(DD), 256, 0, stream>>>(wk, wkb, (long long)DD);
  cast_f32_bf16_kernel<<<gridFor(DD), 256, 0, stream>>>(wv, wvb, (long long)DD);
  cast_f32_bf16_kernel<<<gridFor(DD), 256, 0, stream>>>(wo, wob, (long long)DD);

  // 2) QKV projections (WMMA); each block covers 256 rows x 64 cols
  dim3 gg((S_C + 255) / 256, DIM_C / 64);
  gemm_nt_bf16_kernel<<<gg, 256, 0, stream>>>(xb, wqb, bq, Qp, S_C, DIM_C, DIM_C);
  gemm_nt_bf16_kernel<<<gg, 256, 0, stream>>>(xb, wkb, bk, Kp, S_C, DIM_C, DIM_C);
  gemm_nt_bf16_kernel<<<gg, 256, 0, stream>>>(xb, wvb, bv, Vp, S_C, DIM_C, DIM_C);

  // 3) cache -> bf16 (k: same layout; v: transposed [d][key])
  const size_t CD = (size_t)START_C * DIM_C;
  cast_f32_bf16_kernel<<<gridFor(CD), 256, 0, stream>>>(k_cache, kf, (long long)CD);
  cast_vT_kernel<<<gridFor(CD), 256, 0, stream>>>(v_cache, vT, (long long)START_C);

  // 4) RMSNorm + RoPE + merge new k/v into full buffers
  qkv_post_kernel<<<S_C, 256, 0, stream>>>(Qp, Kp, Vp, nqw, nkw, freqs, qb, kf, vT);

  // 5) flash attention (WMMA)
  dim3 ga((S_C + 15) / 16, NH_C);
  attn_kernel<<<ga, 256, 0, stream>>>(qb, kf, vT, obf);

  // 6) output projection -> d_out (fp32)
  gemm_nt_bf16_kernel<<<gg, 256, 0, stream>>>(obf, wob, bo, out, S_C, DIM_C, DIM_C);
}
